// DNN_Model_33852932227151
// MI455X (gfx1250) — compile-verified, matrix operand order checked by reading, not run on hardware
//
#include <hip/hip_runtime.h>
#include <hip/hip_bf16.h>

// ---------------------------------------------------------------------------
// MI455X reasoning:
//  - MLP (8.3 GFLOP) -> V_WMMA_F32_16X16X32_F16, f32 accum. Tiles staged into
//    double-buffered LDS with GLOBAL_LOAD_ASYNC_TO_LDS_B128 (ASYNCcnt), so
//    next-tile DMA overlaps the current tile's WMMAs.
//  - Weights transposed once during f32->f16 conversion so every staging copy
//    is a contiguous 16B chunk (no per-element transposes, no exec masking).
//  - einsum over T (655 MB) -> bandwidth bound (~28us @ 23.3 TB/s): streaming
//    coalesced loads + prefetch + LDS reduction, one workgroup per sample.
// ---------------------------------------------------------------------------

typedef __attribute__((ext_vector_type(16))) _Float16 v16h;
typedef __attribute__((ext_vector_type(8)))  float    v8f;

#define BB      2048
#define DIN     303
#define KP1     320      // DIN padded to mult of 32
#define N1      1024
#define N2      1024
#define N3      512
#define N4REAL  264
#define NP4     320      // 264 padded to mult of 64
#define NRIS    100
#define INV_THRESH_W 1.0e15f   // 1 / (10^(-120/10)/1000)
#define KC      32

// Async 16B global->LDS copy (CDNA5; tracked by ASYNCcnt).
__device__ __forceinline__ void async_copy_b128(void* lds_ptr, const void* gptr) {
  unsigned lds_off = (unsigned)(size_t)lds_ptr;              // low 32 bits = LDS offset
  asm volatile("global_load_async_to_lds_b128 %0, %1, off"
               :: "v"(lds_off), "v"((unsigned long long)(size_t)gptr)
               : "memory");
}
__device__ __forceinline__ void wait_async() {
  asm volatile("s_wait_asynccnt 0x0" ::: "memory");
}

// ---------------- f32 -> f16 conversion with zero padding -------------------
__global__ void __launch_bounds__(256)
cvt_pad_f16(const float* __restrict__ src, _Float16* __restrict__ dst,
            int rows, int cols, int rowsP, int colsP) {
  long long i = (long long)blockIdx.x * blockDim.x + threadIdx.x;
  long long total = (long long)rowsP * colsP;
  if (i >= total) return;
  int r = (int)(i / colsP);
  int c = (int)(i - (long long)r * colsP);
  float v = (r < rows && c < cols) ? src[(long long)r * cols + c] : 0.0f;
  dst[i] = (_Float16)v;
}

// f32 [K,N] -> f16 transposed [Np,Kp] (weight matrices; done once, amortized)
__global__ void __launch_bounds__(256)
cvt_pad_f16_T(const float* __restrict__ src, _Float16* __restrict__ dst,
              int K, int N, int Kp, int Np) {
  long long i = (long long)blockIdx.x * blockDim.x + threadIdx.x;
  long long total = (long long)Np * Kp;
  if (i >= total) return;
  int n = (int)(i / Kp);
  int k = (int)(i - (long long)n * Kp);
  float v = (n < N && k < K) ? src[(long long)k * N + n] : 0.0f;
  dst[i] = (_Float16)v;
}

// ---------------- WMMA GEMM: out = act(A[M,K] @ W^T + bias) -----------------
// A:  [M,K]  row-major f16, K % 32 == 0
// Wt: [N,K]  (transposed weights), leading dim ldB = K
// Workgroup: 256 threads = 8 waves. Tile: 128(M) x 64(N); wave w -> rows
// [w*16, w*16+16), all 64 cols (4 f32 accumulators). K stepped by 32.
// Exactly 3 unconditional async b128 copies per thread per K-step.
__device__ __forceinline__ void
stage_tile(const _Float16* __restrict__ A, const _Float16* __restrict__ Wt,
           _Float16* lA, _Float16* lB, int tid, int m0, int n0,
           int K, int ldB, int k0)
{
  const int r0 = tid >> 2, c0 = tid & 3;          // chunk tid   (A rows 0..63)
  const int r1 = r0 + 64;                         // chunk tid+256 (A rows 64..127)
  async_copy_b128(lA + r0 * KC + c0 * 8, A  + (long long)(m0 + r0) * K   + k0 + c0 * 8);
  async_copy_b128(lA + r1 * KC + c0 * 8, A  + (long long)(m0 + r1) * K   + k0 + c0 * 8);
  async_copy_b128(lB + r0 * KC + c0 * 8, Wt + (long long)(n0 + r0) * ldB + k0 + c0 * 8);
}

__global__ void __launch_bounds__(256)
gemm_wmma_f16(const _Float16* __restrict__ A,
              const _Float16* __restrict__ Wt, int ldB,
              const float* __restrict__ bias, int biasN,
              _Float16* __restrict__ outH, float* __restrict__ outF,
              int M, int K, int N, int relu)
{
  __shared__ __align__(16) _Float16 lA[2][128 * KC];  // [buf][128 rows][32 k]
  __shared__ __align__(16) _Float16 lB[2][64 * KC];   // [buf][64 cols][32 k]

  const int tid  = threadIdx.x;
  const int lane = tid & 31;
  const int wave = tid >> 5;
  const int m0   = blockIdx.x * 128;
  const int n0   = blockIdx.y * 64;

  v8f acc0 = {}, acc1 = {}, acc2 = {}, acc3 = {};

  const int rowA = lane & 15;            // row within 16-row wave tile
  const int kb   = (lane >> 4) << 3;     // K sub-base: 0 (lanes 0-15) / 8 (16-31)
  const int nT   = K / KC;

  stage_tile(A, Wt, lA[0], lB[0], tid, m0, n0, K, ldB, 0);

  for (int it = 0; it < nT; ++it) {
    const int buf = it & 1;
    wait_async();        // my copies of tile `it` (and any in-flight) landed
    __syncthreads();     // everyone's landed; everyone done reading buf^1

    if (it + 1 < nT)     // overlap next tile's DMA with this tile's WMMAs
      stage_tile(A, Wt, lA[buf ^ 1], lB[buf ^ 1], tid, m0, n0, K, ldB, (it + 1) * KC);

    // A fragment: lane l<16 holds row l, K {0..7,16..23}; l>=16: K {8..15,24..31}
    v16h af;
    {
      const _Float16* ar = lA[buf] + (wave * 16 + rowA) * KC;
      #pragma unroll
      for (int j = 0; j < 8; ++j) {
        af[j]     = ar[kb + j];
        af[8 + j] = ar[16 + kb + j];
      }
    }
    // 4 B fragments (cols t*16 + lane%16), same per-lane K striping.
    #pragma unroll
    for (int t = 0; t < 4; ++t) {
      v16h bf;
      const _Float16* br = lB[buf] + (t * 16 + rowA) * KC;
      #pragma unroll
      for (int j = 0; j < 8; ++j) {
        bf[j]     = br[kb + j];
        bf[8 + j] = br[16 + kb + j];
      }
      if (t == 0) acc0 = __builtin_amdgcn_wmma_f32_16x16x32_f16(false, af, false, bf, (short)0, acc0, false, false);
      if (t == 1) acc1 = __builtin_amdgcn_wmma_f32_16x16x32_f16(false, af, false, bf, (short)0, acc1, false, false);
      if (t == 2) acc2 = __builtin_amdgcn_wmma_f32_16x16x32_f16(false, af, false, bf, (short)0, acc2, false, false);
      if (t == 3) acc3 = __builtin_amdgcn_wmma_f32_16x16x32_f16(false, af, false, bf, (short)0, acc3, false, false);
    }
  }

  // Epilogue. C/D layout: VGPR j -> row j (lanes 0-15) / row 8+j (lanes 16-31),
  // col = lane % 16.
  const int rowBase = m0 + wave * 16 + ((lane >> 4) << 3);
  const int colBase = n0 + (lane & 15);
  #pragma unroll
  for (int j = 0; j < 8; ++j) {
    const int r = rowBase + j;
    float vals[4] = { acc0[j], acc1[j], acc2[j], acc3[j] };
    #pragma unroll
    for (int t = 0; t < 4; ++t) {
      const int col = colBase + t * 16;
      float v = vals[t] + ((col < biasN) ? bias[col] : 0.0f);
      if (relu) v = fmaxf(v, 0.0f);
      const long long oi = (long long)r * N + col;
      if (outH) outH[oi] = (_Float16)v;
      if (outF) outF[oi] = v;
    }
  }
}

// ---------------- Postprocess: phases, F-norm, quad forms, rescale ----------
__global__ void __launch_bounds__(256)
postprocess(const float* __restrict__ theta1, int ldT,
            const float* __restrict__ Tre, const float* __restrict__ Tim,
            float* __restrict__ out)
{
  const int b   = blockIdx.x;
  const int tid = threadIdx.x;
  __shared__ float thA[NRIS], thB[NRIS];
  __shared__ float sF[64];
  __shared__ float red[256];
  __shared__ float sCCC[4];
  __shared__ float sScale;

  const float* t1 = theta1 + (long long)b * ldT;

  // Unit-modulus RIS phases
  if (tid < NRIS) {
    float re = t1[tid], im = t1[NRIS + tid];
    float inv = rsqrtf(re * re + im * im);
    thA[tid] = re * inv;
    thB[tid] = im * inv;
  }
  // Precoder normalization F *= sqrt(2 / ||F||^2)  (thread 100: F1, 101: F2)
  if (tid == 100 || tid == 101) {
    const int off = 200 + (tid - 100) * 32;
    float s = 0.0f;
    for (int k = 0; k < 16; ++k) {
      float re = t1[off + k], im = t1[off + 16 + k];
      s += re * re + im * im;
    }
    float sc = sqrtf(2.0f / s);
    for (int k = 0; k < 32; ++k) sF[(tid - 100) * 32 + k] = t1[off + k] * sc;
  }
  __syncthreads();

  // CCC_c = Re(theta^H T_c theta) / thr  — streaming, bandwidth-bound.
  const long long baseT = (long long)b * 4 * NRIS * NRIS;
  for (int c = 0; c < 4; ++c) {
    const float* tr = Tre + baseT + (long long)c * NRIS * NRIS;
    const float* ti = Tim + baseT + (long long)c * NRIS * NRIS;
    float p = 0.0f;
    for (int i = tid; i < NRIS * NRIS; i += 256) {
      __builtin_prefetch(tr + i + 1024, 0, 0);   // global_prefetch_b8
      __builtin_prefetch(ti + i + 1024, 0, 0);
      int n = i / NRIS;
      int m = i - n * NRIS;
      float am = thA[m], bm = thB[m], an = thA[n], bn = thB[n];
      float r = tr[i], q = ti[i];
      float u = fmaf(r, am, -q * bm);   // Re(T * th_m)
      float v = fmaf(r, bm,  q * am);   // Im(T * th_m)
      p = fmaf(an, u, fmaf(bn, v, p));
    }
    red[tid] = p;
    __syncthreads();
    for (int s = 128; s > 0; s >>= 1) {
      if (tid < s) red[tid] += red[tid + s];
      __syncthreads();
    }
    if (tid == 0) sCCC[c] = red[0] * INV_THRESH_W;
    __syncthreads();
  }

  if (tid == 0) {
    float mx = fmaxf(fmaxf(sCCC[0], sCCC[1]), fmaxf(sCCC[2], sCCC[3]));
    sScale = rsqrtf(fmaxf(mx, 1.0f));
  }
  __syncthreads();

  float* ob = out + (long long)b * 264;
  if (tid < NRIS) {
    ob[tid]        = thA[tid] * sScale;
    ob[NRIS + tid] = thB[tid] * sScale;
  }
  if (tid >= 128 && tid < 192) ob[200 + (tid - 128)] = sF[tid - 128];
}

// ---------------------------------------------------------------------------
extern "C" void kernel_launch(void* const* d_in, const int* in_sizes, int n_in,
                              void* d_out, int out_size, void* d_ws, size_t ws_size,
                              hipStream_t stream) {
  const float* sample1 = (const float*)d_in[0];
  const float* T_real  = (const float*)d_in[1];
  const float* T_imag  = (const float*)d_in[2];
  const float* W1 = (const float*)d_in[3];  const float* b1 = (const float*)d_in[4];
  const float* W2 = (const float*)d_in[5];  const float* b2 = (const float*)d_in[6];
  const float* W3 = (const float*)d_in[7];  const float* b3 = (const float*)d_in[8];
  const float* W4 = (const float*)d_in[9];  const float* b4 = (const float*)d_in[10];
  float* out = (float*)d_out;

  char*  ws  = (char*)d_ws;
  size_t off = 0;
  auto alloc = [&](size_t bytes) -> void* {
    void* p = ws + off;
    off = (off + bytes + 255) & ~(size_t)255;
    return p;
  };

  _Float16* A0   = (_Float16*)alloc((size_t)BB  * KP1 * 2);
  _Float16* W1t  = (_Float16*)alloc((size_t)N1  * KP1 * 2);   // [N1][KP1]
  _Float16* W2t  = (_Float16*)alloc((size_t)N2  * N1  * 2);   // [N2][N1]
  _Float16* W3t  = (_Float16*)alloc((size_t)N3  * N2  * 2);   // [N3][N2]
  _Float16* W4t  = (_Float16*)alloc((size_t)NP4 * N3  * 2);   // [NP4][N3]
  _Float16* H1   = (_Float16*)alloc((size_t)BB  * N1  * 2);
  _Float16* H2   = (_Float16*)alloc((size_t)BB  * N2  * 2);
  _Float16* H3   = (_Float16*)alloc((size_t)BB  * N3  * 2);
  float*    TH1  = (float*)   alloc((size_t)BB  * NP4 * 4);
  (void)ws_size; (void)in_sizes; (void)n_in; (void)out_size;

  const int TPB = 256;
  auto g1d = [](long long n) { return dim3((unsigned)((n + 255) / 256)); };

  // f32 -> f16 conversions; weights transposed to [N][K] for 16B-chunk staging
  cvt_pad_f16  <<<g1d((long long)BB  * KP1), TPB, 0, stream>>>(sample1, A0,  BB,  DIN, BB, KP1);
  cvt_pad_f16_T<<<g1d((long long)N1  * KP1), TPB, 0, stream>>>(W1, W1t, DIN, N1,     KP1, N1);
  cvt_pad_f16_T<<<g1d((long long)N2  * N1 ), TPB, 0, stream>>>(W2, W2t, N1,  N2,     N1,  N2);
  cvt_pad_f16_T<<<g1d((long long)N3  * N2 ), TPB, 0, stream>>>(W3, W3t, N2,  N3,     N2,  N3);
  cvt_pad_f16_T<<<g1d((long long)NP4 * N3 ), TPB, 0, stream>>>(W4, W4t, N3,  N4REAL, N3,  NP4);

  // MLP via WMMA GEMMs (tile 128x64 per 8-wave workgroup, double-buffered
  // async LDS staging)
  gemm_wmma_f16<<<dim3(BB / 128, N1  / 64), TPB, 0, stream>>>(A0, W1t, KP1, b1, N1,     H1, nullptr, BB, KP1, N1,  1);
  gemm_wmma_f16<<<dim3(BB / 128, N2  / 64), TPB, 0, stream>>>(H1, W2t, N1,  b2, N2,     H2, nullptr, BB, N1,  N2,  1);
  gemm_wmma_f16<<<dim3(BB / 128, N3  / 64), TPB, 0, stream>>>(H2, W3t, N2,  b3, N3,     H3, nullptr, BB, N2,  N3,  1);
  gemm_wmma_f16<<<dim3(BB / 128, NP4 / 64), TPB, 0, stream>>>(H3, W4t, N3,  b4, N4REAL, nullptr, TH1, BB, N3, NP4, 0);

  // Phases, precoder norm, quad forms (bandwidth-bound), rescale, pack output
  postprocess<<<dim3(BB), TPB, 0, stream>>>(TH1, NP4, T_real, T_imag, out);
}
